// CharNameTransformer_63754494542648
// MI455X (gfx1250) — compile-verified, matrix operand order checked by reading, not run on hardware
//
#include <hip/hip_runtime.h>
#include <hip/hip_bf16.h>
#include <math.h>

// ---------------- model constants ----------------
#define D_MODEL 256
#define H_HEADS 8
#define DH 32
#define FF_DIM 1024
#define MAX_CHARS 16
#define MAX_WORDS 8
#define N_NAMES 2048
#define N_WORDS 9216
#define CH_WORDS 1152              // 8 chunks of words for stage 1
#define N_CHUNKS (N_WORDS / CH_WORDS)
#define MC_TOK (CH_WORDS * MAX_CHARS)   // 18432 tokens per stage-1 chunk
#define M2_TOK (N_NAMES * MAX_WORDS)    // 16384 tokens for stage 2

#define CDIV(a, b) (((a) + (b) - 1) / (b))

typedef __attribute__((ext_vector_type(16))) _Float16 v16h;
typedef __attribute__((ext_vector_type(8)))  _Float16 v8h;
typedef __attribute__((ext_vector_type(8)))  float    v8f;

// ---------------- weight transpose + f32->f16 ----------------
__global__ void k_wT(const float* __restrict__ W, _Float16* __restrict__ Wt, int K, int N) {
    int i = blockIdx.x * blockDim.x + threadIdx.x;
    if (i >= K * N) return;
    int k = i / N, n = i % N;
    Wt[(long)n * K + k] = (_Float16)W[i];
}

// copy f32 -> (f32 dst, f16 mirror)
__global__ void k_copy_dual(const float* __restrict__ src, float* __restrict__ dst,
                            _Float16* __restrict__ dsth, int n) {
    int i = blockIdx.x * blockDim.x + threadIdx.x;
    if (i >= n) return;
    float v = src[i];
    dst[i] = v;
    dsth[i] = (_Float16)v;
}

__global__ void k_zero_f32(float* __restrict__ p, int n) {
    int i = blockIdx.x * blockDim.x + threadIdx.x;
    if (i < n) p[i] = 0.0f;
}

// ---------------- embedding: x = char_emb[inputs] + pos_emb (f32 + f16 mirror) ----------------
__global__ void k_embed(const int* __restrict__ inputs, const float* __restrict__ char_emb,
                        const float* __restrict__ pos_emb, float* __restrict__ x,
                        _Float16* __restrict__ xh, int words) {
    int tid = blockIdx.x * blockDim.x + threadIdx.x;
    int total = words * MAX_CHARS * D_MODEL;
    if (tid >= total) return;
    int d = tid % D_MODEL;
    int c = (tid / D_MODEL) % MAX_CHARS;
    int w = tid / (D_MODEL * MAX_CHARS);
    int ch = inputs[w * MAX_CHARS + c];
    float v = char_emb[ch * D_MODEL + d] + pos_emb[c * D_MODEL + d];
    x[tid] = v;
    xh[tid] = (_Float16)v;
}

// ---------------- WMMA GEMM: C[M,N] = A[M,K] x Bt[N,K]^T ----------------
// Register-blocked 32x32 C tile per wave (2x2 WMMA tiles): each A/B fragment is
// reused twice -> 8 b128 loads feed 4 v_wmma_f32_16x16x32_f16 per K-step.
template <bool STORE_HALF>
__global__ void k_gemm(const _Float16* __restrict__ A, const _Float16* __restrict__ Bt,
                       void* __restrict__ Cv, int M, int N, int K) {
    int wave = threadIdx.x >> 5;
    int lane = threadIdx.x & 31;
    int tile = blockIdx.x * 8 + wave;          // 8 waves / block, one 32x32 tile each
    int tilesN = N >> 5;
    int tilesM = M >> 5;
    if (tile >= tilesM * tilesN) return;       // wave-uniform: EXEC stays all-ones
    int tm = tile / tilesN, tn = tile % tilesN;
    int l16 = lane & 15;
    int hi  = (lane >> 4) & 1;                 // lane group
    int off1 = hi * 8;                         // lanes 0-15: K{0-7,16-23}; 16-31: K{8-15,24-31}
    const _Float16* aRow0 = A  + (long)(tm * 32 + l16) * K;
    const _Float16* aRow1 = aRow0 + (long)16 * K;
    const _Float16* bRow0 = Bt + (long)(tn * 32 + l16) * K;
    const _Float16* bRow1 = bRow0 + (long)16 * K;
    v8f c00 = {}, c01 = {}, c10 = {}, c11 = {};
    for (int kk = 0; kk < K; kk += 32) {
        v8h a00 = *reinterpret_cast<const v8h*>(aRow0 + kk + off1);
        v8h a01 = *reinterpret_cast<const v8h*>(aRow0 + kk + off1 + 16);
        v8h a10 = *reinterpret_cast<const v8h*>(aRow1 + kk + off1);
        v8h a11 = *reinterpret_cast<const v8h*>(aRow1 + kk + off1 + 16);
        v8h b00 = *reinterpret_cast<const v8h*>(bRow0 + kk + off1);
        v8h b01 = *reinterpret_cast<const v8h*>(bRow0 + kk + off1 + 16);
        v8h b10 = *reinterpret_cast<const v8h*>(bRow1 + kk + off1);
        v8h b11 = *reinterpret_cast<const v8h*>(bRow1 + kk + off1 + 16);
        v16h a0, a1, b0, b1;
#pragma unroll
        for (int i = 0; i < 8; ++i) {
            a0[i] = a00[i]; a0[i + 8] = a01[i];
            a1[i] = a10[i]; a1[i + 8] = a11[i];
            b0[i] = b00[i]; b0[i + 8] = b01[i];
            b1[i] = b10[i]; b1[i + 8] = b11[i];
        }
        c00 = __builtin_amdgcn_wmma_f32_16x16x32_f16(false, a0, false, b0, (short)0, c00, false, false);
        c01 = __builtin_amdgcn_wmma_f32_16x16x32_f16(false, a0, false, b1, (short)0, c01, false, false);
        c10 = __builtin_amdgcn_wmma_f32_16x16x32_f16(false, a1, false, b0, (short)0, c10, false, false);
        c11 = __builtin_amdgcn_wmma_f32_16x16x32_f16(false, a1, false, b1, (short)0, c11, false, false);
    }
    // C/D layout: lane n (0-15) VGPR r = row r, lane n+16 VGPR r = row r+8
    int col0 = tn * 32 + l16;
    int col1 = col0 + 16;
    int row0 = tm * 32 + hi * 8;
    int row1 = row0 + 16;
    if (STORE_HALF) {
        _Float16* C = (_Float16*)Cv;
#pragma unroll
        for (int r = 0; r < 8; ++r) {
            C[(long)(row0 + r) * N + col0] = (_Float16)c00[r];
            C[(long)(row0 + r) * N + col1] = (_Float16)c01[r];
            C[(long)(row1 + r) * N + col0] = (_Float16)c10[r];
            C[(long)(row1 + r) * N + col1] = (_Float16)c11[r];
        }
    } else {
        float* C = (float*)Cv;
#pragma unroll
        for (int r = 0; r < 8; ++r) {
            C[(long)(row0 + r) * N + col0] = c00[r];
            C[(long)(row0 + r) * N + col1] = c01[r];
            C[(long)(row1 + r) * N + col0] = c10[r];
            C[(long)(row1 + r) * N + col1] = c11[r];
        }
    }
}

// ---------------- attention: one thread per (batch, head, query) ----------------
// qkv packed rows of 768 f16: [q(256) | k(256) | v(256)], head h uses cols h*32..h*32+31.
template <int T>
__global__ void k_attn(const _Float16* __restrict__ qkv, _Float16* __restrict__ o,
                       const int* __restrict__ maskInp, int B) {
    int tid = blockIdx.x * blockDim.x + threadIdx.x;
    if (tid >= B * H_HEADS * T) return;
    int t = tid % T;
    int h = (tid / T) % H_HEADS;
    int b = tid / (T * H_HEADS);
    const _Float16* base = qkv + (long)b * T * 768;
    const _Float16* q = base + t * 768 + h * DH;
    float qf[DH];
#pragma unroll
    for (int d = 0; d < DH; ++d) qf[d] = (float)q[d];
    float sc[T];
    float mx = -1e30f;
    for (int s = 0; s < T; ++s) {
        const _Float16* kr = base + s * 768 + 256 + h * DH;
        float acc = 0.0f;
#pragma unroll
        for (int d = 0; d < DH; ++d) acc += qf[d] * (float)kr[d];
        acc *= 0.17677669529663687f;           // 1/sqrt(32)
        if (maskInp && maskInp[b * T + s] == 0) acc = -1e9f;
        sc[s] = acc;
        mx = fmaxf(mx, acc);
    }
    float sum = 0.0f;
    for (int s = 0; s < T; ++s) { sc[s] = __expf(sc[s] - mx); sum += sc[s]; }
    float inv = 1.0f / sum;
    float outv[DH] = {};
    for (int s = 0; s < T; ++s) {
        float p = sc[s] * inv;
        const _Float16* vr = base + s * 768 + 512 + h * DH;
#pragma unroll
        for (int d = 0; d < DH; ++d) outv[d] += p * (float)vr[d];
    }
    _Float16* op = o + (long)(b * T + t) * D_MODEL + h * DH;
#pragma unroll
    for (int d = 0; d < DH; ++d) op[d] = (_Float16)outv[d];
}

// ---------------- fused residual + LayerNorm + f16 mirror, one wave32 per row ----------------
__global__ void k_res_ln(float* __restrict__ x, _Float16* __restrict__ xh,
                         const float* __restrict__ dlt,
                         const float* __restrict__ s, const float* __restrict__ b, int rows) {
    int wave = threadIdx.x >> 5, lane = threadIdx.x & 31;
    int row = blockIdx.x * 8 + wave;
    if (row >= rows) return;
    float* xr = x + (long)row * D_MODEL;
    _Float16* xhr = xh + (long)row * D_MODEL;
    const float* dr = dlt + (long)row * D_MODEL;
    float v[8];
    float sum = 0.0f, sq = 0.0f;
#pragma unroll
    for (int j = 0; j < 8; ++j) {
        float t = xr[lane + 32 * j] + dr[lane + 32 * j];
        v[j] = t; sum += t; sq += t * t;
    }
#pragma unroll
    for (int off = 16; off > 0; off >>= 1) {
        sum += __shfl_xor(sum, off, 32);
        sq  += __shfl_xor(sq,  off, 32);
    }
    float m   = sum * (1.0f / 256.0f);
    float var = sq  * (1.0f / 256.0f) - m * m;
    float r   = rsqrtf(var + 1e-5f);
#pragma unroll
    for (int j = 0; j < 8; ++j) {
        int d = lane + 32 * j;
        float y = (v[j] - m) * r * s[d] + b[d];
        xr[d] = y;
        xhr[d] = (_Float16)y;
    }
}

// ---------------- gelu (tanh approx, jax default) in-place on f16 ----------------
__global__ void k_gelu(_Float16* __restrict__ h, int n) {
    int i = blockIdx.x * blockDim.x + threadIdx.x;
    if (i >= n) return;
    float x = (float)h[i];
    float c = 0.7978845608028654f * (x + 0.044715f * x * x * x);
    h[i] = (_Float16)(0.5f * x * (1.0f + tanhf(c)));
}

// ---------------- masked mean over chars -> word_vec ----------------
__global__ void k_word_pool(const float* __restrict__ x, const int* __restrict__ inputs,
                            float* __restrict__ wv, int words) {
    int tid = blockIdx.x * blockDim.x + threadIdx.x;
    if (tid >= words * D_MODEL) return;
    int d = tid % D_MODEL, w = tid / D_MODEL;
    float s = 0.0f; float cnt = 0.0f;
    for (int c = 0; c < MAX_CHARS; ++c) {
        if (inputs[w * MAX_CHARS + c] != 0) {
            s += x[(long)(w * MAX_CHARS + c) * D_MODEL + d];
            cnt += 1.0f;
        }
    }
    wv[tid] = s / cnt;
}

// ---------------- scatter word_vec into name_emb ----------------
__global__ void k_scatter(const float* __restrict__ wv, const int* __restrict__ w2n,
                          const int* __restrict__ wpos, float* __restrict__ ne) {
    int tid = blockIdx.x * blockDim.x + threadIdx.x;
    if (tid >= N_WORDS * D_MODEL) return;
    int d = tid % D_MODEL, w = tid / D_MODEL;
    ne[(long)(w2n[w] * MAX_WORDS + wpos[w]) * D_MODEL + d] = wv[tid];
}

// ---------------- elementwise-masked mean over word slots ----------------
__global__ void k_name_pool(const float* __restrict__ y, const float* __restrict__ ne,
                            float* __restrict__ out) {
    int tid = blockIdx.x * blockDim.x + threadIdx.x;
    if (tid >= N_NAMES * D_MODEL) return;
    int d = tid % D_MODEL, n = tid / D_MODEL;
    float num = 0.0f, den = 0.0f;
    for (int j = 0; j < MAX_WORDS; ++j) {
        long idx = (long)(n * MAX_WORDS + j) * D_MODEL + d;
        if (ne[idx] != 0.0f) { num += y[idx]; den += 1.0f; }
    }
    out[tid] = num / den;
}

// ================= host orchestration =================
extern "C" void kernel_launch(void* const* d_in, const int* in_sizes, int n_in,
                              void* d_out, int out_size, void* d_ws, size_t ws_size,
                              hipStream_t stream) {
    const int*   inputs   = (const int*)d_in[0];
    const int*   w2n      = (const int*)d_in[1];
    const int*   wpos     = (const int*)d_in[2];
    const float* char_emb = (const float*)d_in[3];
    const float* pos_emb  = (const float*)d_in[4];
    const float* Wqkv_src[2] = { (const float*)d_in[5],  (const float*)d_in[13] };
    const float* Wo_src[2]   = { (const float*)d_in[6],  (const float*)d_in[14] };
    const float* ln1s[2]     = { (const float*)d_in[7],  (const float*)d_in[15] };
    const float* ln1b[2]     = { (const float*)d_in[8],  (const float*)d_in[16] };
    const float* W1_src[2]   = { (const float*)d_in[9],  (const float*)d_in[17] };
    const float* W2_src[2]   = { (const float*)d_in[10], (const float*)d_in[18] };
    const float* ln2s[2]     = { (const float*)d_in[11], (const float*)d_in[19] };
    const float* ln2b[2]     = { (const float*)d_in[12], (const float*)d_in[20] };
    float* out = (float*)d_out;

    // ---- workspace carve-up (~146 MB total) ----
    char* wsb = (char*)d_ws;
    size_t off = 0;
    auto carve = [&](size_t bytes) {
        size_t r = off;
        off = (off + bytes + 255) & ~(size_t)255;
        return (void*)(wsb + r);
    };
    const size_t SET_H = 786432; // halves per converted layer-set
    _Float16* wts      = (_Float16*)carve(4 * SET_H * sizeof(_Float16));
    float*    word_vec = (float*)carve((size_t)N_WORDS * D_MODEL * 4);
    float*    name_emb = (float*)carve((size_t)N_NAMES * MAX_WORDS * D_MODEL * 4);
    float*    xf32     = (float*)carve((size_t)MC_TOK * D_MODEL * 4);
    _Float16* xh       = (_Float16*)carve((size_t)MC_TOK * D_MODEL * 2);
    _Float16* qkvh     = (_Float16*)carve((size_t)MC_TOK * 768 * 2);
    float*    delta    = (float*)carve((size_t)MC_TOK * D_MODEL * 4);
    _Float16* h1h      = (_Float16*)carve((size_t)MC_TOK * FF_DIM * 2);
    (void)ws_size; (void)n_in; (void)in_sizes; (void)out_size;

    // ---- convert & transpose all weights to f16 [N][K] (stays hot in 192MB L2) ----
    for (int e = 0; e < 2; ++e) {
        for (int l = 0; l < 2; ++l) {
            _Float16* base = wts + (size_t)(e * 2 + l) * SET_H;
            k_wT<<<CDIV(256 * 768, 256), 256, 0, stream>>>(Wqkv_src[e] + (size_t)l * 256 * 768, base, 256, 768);
            k_wT<<<CDIV(256 * 256, 256), 256, 0, stream>>>(Wo_src[e] + (size_t)l * 256 * 256, base + 196608, 256, 256);
            k_wT<<<CDIV(256 * 1024, 256), 256, 0, stream>>>(W1_src[e] + (size_t)l * 256 * 1024, base + 262144, 256, 1024);
            k_wT<<<CDIV(1024 * 256, 256), 256, 0, stream>>>(W2_src[e] + (size_t)l * 1024 * 256, base + 524288, 1024, 256);
        }
    }

    auto gemm = [&](const _Float16* A, const _Float16* Bt, void* C, int M, int N, int K, bool half_out) {
        int tiles = (M >> 5) * (N >> 5);       // 32x32 tiles
        int blocks = CDIV(tiles, 8);
        if (half_out) k_gemm<true><<<blocks, 256, 0, stream>>>(A, Bt, C, M, N, K);
        else          k_gemm<false><<<blocks, 256, 0, stream>>>(A, Bt, C, M, N, K);
    };

    // one encoder layer on residual stream x (f32) with f16 mirror xh kept in sync
    auto run_layer = [&](float* x, int M, int T, const int* maskInp, int e, int l) {
        const _Float16* base = wts + (size_t)(e * 2 + l) * SET_H;
        const _Float16* WqkvT = base;
        const _Float16* WoT   = base + 196608;
        const _Float16* W1T   = base + 262144;
        const _Float16* W2T   = base + 524288;
        gemm(xh, WqkvT, qkvh, M, 768, 256, true);
        int athreads = M * H_HEADS;                     // (M/T)*H*T
        if (T == 16) k_attn<16><<<CDIV(athreads, 256), 256, 0, stream>>>(qkvh, xh, maskInp, M / T);
        else         k_attn<8><<<CDIV(athreads, 256), 256, 0, stream>>>(qkvh, xh, maskInp, M / T);
        gemm(xh, WoT, delta, M, 256, 256, false);
        k_res_ln<<<CDIV(M, 8), 256, 0, stream>>>(x, xh, delta, ln1s[e] + l * 256, ln1b[e] + l * 256, M);
        gemm(xh, W1T, h1h, M, 1024, 256, true);
        k_gelu<<<CDIV(M * FF_DIM, 256), 256, 0, stream>>>(h1h, M * FF_DIM);
        gemm(h1h, W2T, delta, M, 256, 1024, false);
        k_res_ln<<<CDIV(M, 8), 256, 0, stream>>>(x, xh, delta, ln2s[e] + l * 256, ln2b[e] + l * 256, M);
    };

    // ---- stage 1: word encoder, chunked over words ----
    for (int ck = 0; ck < N_CHUNKS; ++ck) {
        int w0 = ck * CH_WORDS;
        const int* inp_ck = inputs + (size_t)w0 * MAX_CHARS;
        int etotal = CH_WORDS * MAX_CHARS * D_MODEL;
        k_embed<<<CDIV(etotal, 256), 256, 0, stream>>>(inp_ck, char_emb, pos_emb, xf32, xh, CH_WORDS);
        run_layer(xf32, MC_TOK, MAX_CHARS, inp_ck, 0, 0);
        run_layer(xf32, MC_TOK, MAX_CHARS, inp_ck, 0, 1);
        k_word_pool<<<CDIV(CH_WORDS * D_MODEL, 256), 256, 0, stream>>>(
            xf32, inp_ck, word_vec + (size_t)w0 * D_MODEL, CH_WORDS);
    }

    // ---- scatter into name_emb (padded slots zero) ----
    int ne_n = N_NAMES * MAX_WORDS * D_MODEL;
    k_zero_f32<<<CDIV(ne_n, 256), 256, 0, stream>>>(name_emb, ne_n);
    k_scatter<<<CDIV(N_WORDS * D_MODEL, 256), 256, 0, stream>>>(word_vec, w2n, wpos, name_emb);

    // ---- stage 2: name encoder (keep name_emb intact for np2 mask) ----
    k_copy_dual<<<CDIV(ne_n, 256), 256, 0, stream>>>(name_emb, xf32, xh, ne_n);
    run_layer(xf32, M2_TOK, MAX_WORDS, nullptr, 1, 0);
    run_layer(xf32, M2_TOK, MAX_WORDS, nullptr, 1, 1);

    // ---- elementwise-masked mean over slots -> output ----
    k_name_pool<<<CDIV(N_NAMES * D_MODEL, 256), 256, 0, stream>>>(xf32, name_emb, out);
}
